// CriticNet_48773648613784
// MI455X (gfx1250) — matrix operand order repaired and checked by reading, not compiled, not run on hardware
//
#include <hip/hip_runtime.h>
#include <hip/hip_bf16.h>

#define B_   32
#define N_   8192
#define E_   131072
#define C_   64

typedef __attribute__((ext_vector_type(16))) _Float16 v16h;
typedef __attribute__((ext_vector_type(8)))  float    v8f;

// ---- monotonic float<->uint key mapping (for radix top-k select & float atomicMax) ----
__device__ __forceinline__ unsigned ordKey(float f) {
  unsigned u = __float_as_uint(f);
  return u ^ ((u >> 31) ? 0xFFFFFFFFu : 0x80000000u);
}
__device__ __forceinline__ float ordDecode(unsigned u) {
  unsigned bits = (u & 0x80000000u) ? (u ^ 0x80000000u) : ~u;
  return __uint_as_float(bits);
}

// =================== SAGE layer 1 (2 -> 64), edge agg + combine ===================
__global__ void k_sage1_edge(const float* __restrict__ x, const int* __restrict__ ei,
                             float* __restrict__ s2, float* __restrict__ cnt) {
  long long idx = (long long)blockIdx.x * blockDim.x + threadIdx.x;
  if (idx >= (long long)B_ * E_) return;
  int b = (int)(idx / E_), e = (int)(idx % E_);
  const int* src = ei + (size_t)b * 2 * E_;
  const int* dst = src + E_;
  int s = src[e], d = dst[e];
  const float* xp = x + ((size_t)b * N_ + s) * 2;
  float* sp = s2 + ((size_t)b * N_ + d) * 2;
  atomicAdd(sp + 0, xp[0]);
  atomicAdd(sp + 1, xp[1]);
  atomicAdd(cnt + (size_t)b * N_ + d, 1.0f);
}

__global__ void k_sage1_combine(const float* __restrict__ x, const float* __restrict__ s2,
                                const float* __restrict__ cnt,
                                const float* __restrict__ Wl, const float* __restrict__ bl,
                                const float* __restrict__ Wr, float* __restrict__ h) {
  long long idx = (long long)blockIdx.x * blockDim.x + threadIdx.x;
  if (idx >= (long long)B_ * N_ * C_) return;
  int c = (int)(idx % C_);
  long long bi = idx / C_;                 // flat b*N + i (level 0 uses all N rows)
  float inv = 1.f / fmaxf(cnt[bi], 1.f);
  float m0 = s2[bi * 2 + 0] * inv, m1 = s2[bi * 2 + 1] * inv;
  float x0 = x[bi * 2 + 0],       x1 = x[bi * 2 + 1];
  float v = m0 * Wl[c * 2 + 0] + m1 * Wl[c * 2 + 1] + bl[c]
          + x0 * Wr[c * 2 + 0] + x1 * Wr[c * 2 + 1];
  h[bi * C_ + c] = fmaxf(v, 0.f);
}

// =================== top-k pooling pieces ===================
__global__ void k_score(const float* __restrict__ h, const float* __restrict__ p,
                        float* __restrict__ score, int n) {
  long long idx = (long long)blockIdx.x * blockDim.x + threadIdx.x;
  if (idx >= (long long)B_ * n) return;
  int b = (int)(idx / n), i = (int)(idx % n);
  const float* hp = h + ((size_t)b * N_ + i) * C_;
  float dot = 0.f, nn = 0.f;
#pragma unroll
  for (int c = 0; c < C_; ++c) { dot += hp[c] * p[c]; nn += p[c] * p[c]; }
  score[(size_t)b * N_ + i] = dot * rsqrtf(nn);
}

// one block per graph: 4-pass MSB radix select of the k-th largest key
__global__ void k_radix_select(const float* __restrict__ score, int n, int k,
                               unsigned* __restrict__ thr, unsigned* __restrict__ quota) {
  __shared__ unsigned hist[256];
  __shared__ unsigned s_prefix, s_want;
  int b = blockIdx.x, t = threadIdx.x;
  const float* sp = score + (size_t)b * N_;
  if (t == 0) { s_prefix = 0u; s_want = (unsigned)k; }
  __syncthreads();
  for (int pass = 3; pass >= 0; --pass) {
    hist[t] = 0u;
    __syncthreads();
    unsigned shift = (unsigned)pass * 8u;
    unsigned pmask = (pass == 3) ? 0u : (0xFFFFFFFFu << ((pass + 1) * 8));
    unsigned pref = s_prefix;
    for (int i = t; i < n; i += 256) {
      unsigned u = ordKey(sp[i]);
      if ((u & pmask) == pref) atomicAdd(&hist[(u >> shift) & 255u], 1u);
    }
    __syncthreads();
    if (t == 0) {
      unsigned want = s_want, cum = 0; int sel = 0;
      for (int bin = 255; bin >= 0; --bin) {
        if (cum + hist[bin] >= want) { sel = bin; break; }
        cum += hist[bin];
      }
      s_want = want - cum;                       // # of ==threshold keys to keep
      s_prefix = pref | ((unsigned)sel << shift);
    }
    __syncthreads();
  }
  if (t == 0) { thr[b] = s_prefix; quota[b] = s_want; }
}

__global__ void k_assign_ranks(const float* __restrict__ score, const unsigned* __restrict__ thr,
                               const unsigned* __restrict__ quota, unsigned* __restrict__ ctr,
                               int* __restrict__ newpos, int n) {
  long long idx = (long long)blockIdx.x * blockDim.x + threadIdx.x;
  if (idx >= (long long)B_ * n) return;
  int b = (int)(idx / n), i = (int)(idx % n);
  unsigned u = ordKey(score[(size_t)b * N_ + i]);
  unsigned T = thr[b];
  int pos = -1;
  if (u > T) pos = (int)atomicAdd(&ctr[b * 2 + 0], 1u);
  else if (u == T) {
    unsigned tie = atomicAdd(&ctr[b * 2 + 1], 1u);
    if (tie < quota[b]) pos = (int)atomicAdd(&ctr[b * 2 + 0], 1u);
  }
  newpos[(size_t)b * N_ + i] = pos;
}

__global__ void k_gather_rows(const float* __restrict__ h, const float* __restrict__ score,
                              const int* __restrict__ newpos, float* __restrict__ xn, int n) {
  long long idx = (long long)blockIdx.x * blockDim.x + threadIdx.x;
  if (idx >= (long long)B_ * n * C_) return;
  int c = (int)(idx % C_);
  long long t2 = idx / C_;
  int b = (int)(t2 / n), i = (int)(t2 % n);
  int p = newpos[(size_t)b * N_ + i];
  if (p < 0) return;
  float tv = tanhf(score[(size_t)b * N_ + i]);
  xn[((size_t)b * N_ + p) * C_ + c] = h[((size_t)b * N_ + i) * C_ + c] * tv;
}

__global__ void k_edge_remap(const int* __restrict__ src_in, const int* __restrict__ dst_in,
                             int estride, const float* __restrict__ m_in,
                             const int* __restrict__ newpos,
                             int* __restrict__ src_out, int* __restrict__ dst_out,
                             float* __restrict__ m_out) {
  long long idx = (long long)blockIdx.x * blockDim.x + threadIdx.x;
  if (idx >= (long long)B_ * E_) return;
  int b = (int)(idx / E_), e = (int)(idx % E_);
  int s = src_in[(size_t)b * estride + e];
  int d = dst_in[(size_t)b * estride + e];
  float m = m_in ? m_in[(size_t)b * E_ + e] : 1.f;
  int ns = newpos[(size_t)b * N_ + s];
  int nd = newpos[(size_t)b * N_ + d];
  bool keep = (m != 0.f) && (ns >= 0) && (nd >= 0);
  src_out[(size_t)b * E_ + e] = ns > 0 ? ns : 0;
  dst_out[(size_t)b * E_ + e] = nd > 0 ? nd : 0;
  m_out[(size_t)b * E_ + e] = keep ? 1.f : 0.f;
}

__global__ void k_stats(const float* __restrict__ xn, int k,
                        unsigned* __restrict__ maxbuf, float* __restrict__ sumbuf) {
  long long idx = (long long)blockIdx.x * blockDim.x + threadIdx.x;
  if (idx >= (long long)B_ * k * C_) return;
  int c = (int)(idx % C_);
  long long t2 = idx / C_;
  int b = (int)(t2 / k), i = (int)(t2 % k);
  float v = xn[((size_t)b * N_ + i) * C_ + c];
  atomicMax(&maxbuf[b * C_ + c], ordKey(v));
  atomicAdd(&sumbuf[b * C_ + c], v);
}

__global__ void k_feats_accum(const unsigned* __restrict__ maxbuf, const float* __restrict__ sumbuf,
                              int k, float* __restrict__ feats) {
  int idx = blockIdx.x * blockDim.x + threadIdx.x;
  if (idx >= B_ * C_) return;
  int b = idx / C_, c = idx % C_;
  feats[b * 128 + c]       += ordDecode(maxbuf[b * C_ + c]);
  feats[b * 128 + 64 + c]  += sumbuf[b * C_ + c] / (float)k;
}

// =================== 64-wide segment ops ===================
__global__ void k_edge_cnt(const int* __restrict__ dst_base, int estride,
                           const float* __restrict__ m_in, float* __restrict__ cnt) {
  long long idx = (long long)blockIdx.x * blockDim.x + threadIdx.x;
  if (idx >= (long long)B_ * E_) return;
  int b = (int)(idx / E_), e = (int)(idx % E_);
  float m = m_in ? m_in[(size_t)b * E_ + e] : 1.f;
  if (m == 0.f) return;
  int d = dst_base[(size_t)b * estride + e];
  atomicAdd(&cnt[(size_t)b * N_ + d], m);
}

__global__ void k_edge_gather_sum(const float* __restrict__ h, const int* __restrict__ src,
                                  const int* __restrict__ dst, int estride,
                                  const float* __restrict__ m_in, float* __restrict__ s) {
  long long idx = (long long)blockIdx.x * blockDim.x + threadIdx.x;
  if (idx >= (long long)B_ * E_ * C_) return;
  int c = (int)(idx % C_);
  long long t2 = idx / C_;
  int b = (int)(t2 / E_), e = (int)(t2 % E_);
  float m = m_in[(size_t)b * E_ + e];
  if (m == 0.f) return;
  int si = src[(size_t)b * estride + e], di = dst[(size_t)b * estride + e];
  atomicAdd(&s[((size_t)b * N_ + di) * C_ + c], h[((size_t)b * N_ + si) * C_ + c]);
}

__global__ void k_mean_div(float* __restrict__ s, const float* __restrict__ cnt, int n) {
  long long idx = (long long)blockIdx.x * blockDim.x + threadIdx.x;
  if (idx >= (long long)B_ * n * C_) return;
  int c = (int)(idx % C_);
  long long t2 = idx / C_;
  int b = (int)(t2 / n), i = (int)(t2 % n);
  s[((size_t)b * N_ + i) * C_ + c] /= fmaxf(cnt[(size_t)b * N_ + i], 1.f);
}

__global__ void k_dis(const float* __restrict__ cnt, float* __restrict__ dis, int n) {
  long long idx = (long long)blockIdx.x * blockDim.x + threadIdx.x;
  if (idx >= (long long)B_ * n) return;
  int b = (int)(idx / n), i = (int)(idx % n);
  dis[(size_t)b * N_ + i] = rsqrtf(cnt[(size_t)b * N_ + i] + 1.f);
}

__global__ void k_edge_gather_norm(const float* __restrict__ hp, const int* __restrict__ src,
                                   const int* __restrict__ dst, int estride,
                                   const float* __restrict__ m_in, const float* __restrict__ dis,
                                   float* __restrict__ agg) {
  long long idx = (long long)blockIdx.x * blockDim.x + threadIdx.x;
  if (idx >= (long long)B_ * E_ * C_) return;
  int c = (int)(idx % C_);
  long long t2 = idx / C_;
  int b = (int)(t2 / E_), e = (int)(t2 % E_);
  float m = m_in[(size_t)b * E_ + e];
  if (m == 0.f) return;
  int si = src[(size_t)b * estride + e], di = dst[(size_t)b * estride + e];
  float nrm = dis[(size_t)b * N_ + si] * dis[(size_t)b * N_ + di] * m;
  atomicAdd(&agg[((size_t)b * N_ + di) * C_ + c],
            hp[((size_t)b * N_ + si) * C_ + c] * nrm);
}

__global__ void k_gcn_combine(float* __restrict__ hp, const float* __restrict__ agg,
                              const float* __restrict__ dis, const float* __restrict__ bg, int n) {
  long long idx = (long long)blockIdx.x * blockDim.x + threadIdx.x;
  if (idx >= (long long)B_ * n * C_) return;
  int c = (int)(idx % C_);
  long long t2 = idx / C_;
  int b = (int)(t2 / n), i = (int)(t2 % n);
  float dd = dis[(size_t)b * N_ + i];
  size_t o = ((size_t)b * N_ + i) * C_ + c;
  hp[o] = fmaxf(agg[o] + hp[o] * dd * dd + bg[c], 0.f);
}

// =================== WMMA GEMM: OUT(n x 64) = act(A1@W1^T [+ A2@W2^T] + bias) ===================
// A tile 16x32 f16 (ISA layout): lane m16 = row, hi half selects K groups {hi*8..+7, 16+hi*8..+7}
__device__ __forceinline__ v16h wmma_loadA(const float* A, size_t abase, int i0, int kk,
                                           int m16, int hi) {
  v16h a;
  const float* ap = A + abase + (size_t)(i0 + m16) * C_ + kk + hi * 8;
#pragma unroll
  for (int e2 = 0; e2 < 8; ++e2) { a[e2] = (_Float16)ap[e2]; a[8 + e2] = (_Float16)ap[16 + e2]; }
  return a;
}
// B tile 32x16 f16 (B[k,j] = W[j,k]): lane m16 = column j, hi half selects K groups {hi*8..+7, 16+hi*8..+7}
__device__ __forceinline__ v16h wmma_loadB(const float* W, int j0, int kk, int m16, int hi) {
  v16h bm;
  const float* wp = W + (size_t)(j0 + m16) * C_ + kk + hi * 8;
#pragma unroll
  for (int e2 = 0; e2 < 8; ++e2) { bm[e2] = (_Float16)wp[e2]; bm[8 + e2] = (_Float16)wp[16 + e2]; }
  return bm;
}

__global__ void k_gemm64(const float* __restrict__ A1, const float* __restrict__ W1,
                         const float* __restrict__ A2, const float* __restrict__ W2,
                         const float* __restrict__ bias, float* __restrict__ OUT,
                         int n, int relu) {
  const int lane = threadIdx.x & 31;
  const int wave = threadIdx.x >> 5;            // 4 waves / block
  const int b = blockIdx.z;
  const int i0 = (blockIdx.x * 4 + wave) * 16;  // 16-row output tile
  const int j0 = blockIdx.y * 16;               // 16-col output tile
  const int m16 = lane & 15, hi = lane >> 4;
  const size_t abase = (size_t)b * N_ * C_;

  v8f acc = {};
#pragma unroll
  for (int kk = 0; kk < C_; kk += 32) {
    v16h a = wmma_loadA(A1, abase, i0, kk, m16, hi);
    v16h w = wmma_loadB(W1, j0, kk, m16, hi);
    acc = __builtin_amdgcn_wmma_f32_16x16x32_f16(false, a, false, w, (short)0, acc, false, false);
  }
  if (A2) {  // uniform branch: EXEC stays all-ones for WMMA
#pragma unroll
    for (int kk = 0; kk < C_; kk += 32) {
      v16h a = wmma_loadA(A2, abase, i0, kk, m16, hi);
      v16h w = wmma_loadB(W2, j0, kk, m16, hi);
      acc = __builtin_amdgcn_wmma_f32_16x16x32_f16(false, a, false, w, (short)0, acc, false, false);
    }
  }
  float bv = bias ? bias[j0 + m16] : 0.f;
#pragma unroll
  for (int r = 0; r < 8; ++r) {                 // D layout: lane col = m16, row = hi*8 + r
    int row = i0 + hi * 8 + r;
    float v = acc[r] + bv;
    if (relu) v = fmaxf(v, 0.f);
    OUT[abase + (size_t)row * C_ + (j0 + m16)] = v;
  }
  (void)n;
}

// =================== final MLP head (136 -> 128 -> 100 -> 1), one block per graph ===================
__global__ void k_head(const float* __restrict__ feats, const float* __restrict__ action,
                       const float* __restrict__ W1, const float* __restrict__ b1,
                       const float* __restrict__ W2, const float* __restrict__ b2,
                       const float* __restrict__ W3, const float* __restrict__ b3,
                       float* __restrict__ out) {
  __shared__ float comb[136];
  __shared__ float h1s[128];
  __shared__ float h2s[100];
  __shared__ float red[128];
  int b = blockIdx.x, t = threadIdx.x;
  comb[t] = feats[b * 128 + t];
  if (t < 8) comb[128 + t] = action[b * 8 + t];
  __syncthreads();
  float a1 = b1[t];
  for (int kx = 0; kx < 136; ++kx) a1 += W1[t * 136 + kx] * comb[kx];
  h1s[t] = fmaxf(a1, 0.f);
  __syncthreads();
  if (t < 100) {
    float a2 = b2[t];
    for (int kx = 0; kx < 128; ++kx) a2 += W2[t * 128 + kx] * h1s[kx];
    h2s[t] = fmaxf(a2, 0.f);
  }
  __syncthreads();
  red[t] = (t < 100) ? W3[t] * h2s[t] : 0.f;
  __syncthreads();
  for (int s = 64; s > 0; s >>= 1) { if (t < s) red[t] += red[t + s]; __syncthreads(); }
  if (t == 0) out[b] = red[0] + b3[0];
}

// =================== host orchestration ===================
extern "C" void kernel_launch(void* const* d_in, const int* in_sizes, int n_in,
                              void* d_out, int out_size, void* d_ws, size_t ws_size,
                              hipStream_t stream) {
  (void)in_sizes; (void)n_in; (void)out_size; (void)ws_size;
  const float* x      = (const float*)d_in[0];
  const int*   ei     = (const int*)  d_in[1];
  const float* action = (const float*)d_in[2];
  const float* Wl1 = (const float*)d_in[3];  const float* bl1 = (const float*)d_in[4];
  const float* Wr1 = (const float*)d_in[5];
  const float* Wl2 = (const float*)d_in[6];  const float* bl2 = (const float*)d_in[7];
  const float* Wr2 = (const float*)d_in[8];
  const float* Wg3 = (const float*)d_in[9];  const float* bg3 = (const float*)d_in[10];
  const float* Wg4 = (const float*)d_in[11]; const float* bg4 = (const float*)d_in[12];
  const float* p1 = (const float*)d_in[13];  const float* p2 = (const float*)d_in[14];
  const float* p3 = (const float*)d_in[15];  const float* p4 = (const float*)d_in[16];
  const float* W1 = (const float*)d_in[17];  const float* b1 = (const float*)d_in[18];
  const float* W2 = (const float*)d_in[19];  const float* b2 = (const float*)d_in[20];
  const float* W3 = (const float*)d_in[21];  const float* b3 = (const float*)d_in[22];
  float* out = (float*)d_out;

  char* base = (char*)d_ws;
  size_t off = 0;
  auto carve = [&](size_t bytes) -> void* {
    void* p = base + off; off += (bytes + 255) & ~(size_t)255; return p;
  };
  const size_t HBYTES = (size_t)B_ * N_ * C_ * sizeof(float);
  float* hX = (float*)carve(HBYTES);
  float* hY = (float*)carve(HBYTES);
  float* hZ = (float*)carve(HBYTES);
  float* s2d   = (float*)carve((size_t)B_ * N_ * 2 * 4);
  float* cnt   = (float*)carve((size_t)B_ * N_ * 4);
  float* dis   = (float*)carve((size_t)B_ * N_ * 4);
  float* score = (float*)carve((size_t)B_ * N_ * 4);
  int*   newpos= (int*)  carve((size_t)B_ * N_ * 4);
  int*   srcA  = (int*)  carve((size_t)B_ * E_ * 4);
  int*   dstA  = (int*)  carve((size_t)B_ * E_ * 4);
  float* mA    = (float*)carve((size_t)B_ * E_ * 4);
  int*   srcB  = (int*)  carve((size_t)B_ * E_ * 4);
  int*   dstB  = (int*)  carve((size_t)B_ * E_ * 4);
  float* mB    = (float*)carve((size_t)B_ * E_ * 4);
  unsigned* maxbuf = (unsigned*)carve((size_t)B_ * C_ * 4);
  float*    sumbuf = (float*)   carve((size_t)B_ * C_ * 4);
  float*    feats  = (float*)   carve((size_t)B_ * 128 * 4);
  unsigned* thr    = (unsigned*)carve((size_t)B_ * 4);
  unsigned* quota  = (unsigned*)carve((size_t)B_ * 4);
  unsigned* ctr    = (unsigned*)carve((size_t)B_ * 2 * 4);

  auto cd = [](long long a, long long b) { return (unsigned)((a + b - 1) / b); };
  const long long BE = (long long)B_ * E_;

  hipMemsetAsync(feats, 0, (size_t)B_ * 128 * 4, stream);

  // ---- Layer 1: SAGE(2 -> 64) over full graphs ----
  hipMemsetAsync(s2d, 0, (size_t)B_ * N_ * 2 * 4, stream);
  hipMemsetAsync(cnt, 0, (size_t)B_ * N_ * 4, stream);
  k_sage1_edge<<<cd(BE, 256), 256, 0, stream>>>(x, ei, s2d, cnt);
  k_sage1_combine<<<cd((long long)B_ * N_ * C_, 256), 256, 0, stream>>>(x, s2d, cnt, Wl1, bl1, Wr1, hX);

  auto pool = [&](const float* h, int n, int k, const float* pvec,
                  const int* src_in, const int* dst_in, int estride, const float* m_in,
                  float* xn, int* src_out, int* dst_out, float* m_out) {
    long long tn = (long long)B_ * n;
    k_score<<<cd(tn, 256), 256, 0, stream>>>(h, pvec, score, n);
    k_radix_select<<<B_, 256, 0, stream>>>(score, n, k, thr, quota);
    hipMemsetAsync(ctr, 0, (size_t)B_ * 2 * 4, stream);
    k_assign_ranks<<<cd(tn, 256), 256, 0, stream>>>(score, thr, quota, ctr, newpos, n);
    k_gather_rows<<<cd(tn * C_, 256), 256, 0, stream>>>(h, score, newpos, xn, n);
    if (src_out)
      k_edge_remap<<<cd(BE, 256), 256, 0, stream>>>(src_in, dst_in, estride, m_in, newpos,
                                                    src_out, dst_out, m_out);
    hipMemsetAsync(maxbuf, 0, (size_t)B_ * C_ * 4, stream);  // 0 == ordKey minimum
    hipMemsetAsync(sumbuf, 0, (size_t)B_ * C_ * 4, stream);
    k_stats<<<cd((long long)B_ * k * C_, 256), 256, 0, stream>>>(xn, k, maxbuf, sumbuf);
    k_feats_accum<<<cd((long long)B_ * C_, 256), 256, 0, stream>>>(maxbuf, sumbuf, k, feats);
  };

  const int k1 = N_ / 2, k2 = N_ / 4, k3 = N_ / 8, k4 = N_ / 16;

  // pool1: hX (n=N) -> hZ ; input edges -> (srcA,dstA,mA)
  pool(hX, N_, k1, p1, ei, ei + E_, 2 * E_, nullptr, hZ, srcA, dstA, mA);

  // ---- Layer 2: SAGE(64 -> 64) on hZ (k1 rows), edges A ----
  hipMemsetAsync(hY, 0, HBYTES, stream);
  hipMemsetAsync(cnt, 0, (size_t)B_ * N_ * 4, stream);
  k_edge_cnt<<<cd(BE, 256), 256, 0, stream>>>(dstA, E_, mA, cnt);
  k_edge_gather_sum<<<cd(BE * C_, 256), 256, 0, stream>>>(hZ, srcA, dstA, E_, mA, hY);
  k_mean_div<<<cd((long long)B_ * k1 * C_, 256), 256, 0, stream>>>(hY, cnt, k1);
  k_gemm64<<<dim3(k1 / 64, 4, B_), 128, 0, stream>>>(hY, Wl2, hZ, Wr2, bl2, hX, k1, 1);

  // pool2: hX (n=k1) -> hZ ; edges A -> B
  pool(hX, k1, k2, p2, srcA, dstA, E_, mA, hZ, srcB, dstB, mB);

  auto gcn = [&](const float* h, int n, const float* W, const float* bg,
                 const int* src, const int* dst, const float* m) {
    k_gemm64<<<dim3(n / 64, 4, B_), 128, 0, stream>>>(h, W, nullptr, nullptr, nullptr, hX, n, 0);
    hipMemsetAsync(cnt, 0, (size_t)B_ * N_ * 4, stream);
    k_edge_cnt<<<cd(BE, 256), 256, 0, stream>>>(dst, E_, m, cnt);
    k_dis<<<cd((long long)B_ * n, 256), 256, 0, stream>>>(cnt, dis, n);
    hipMemsetAsync(hY, 0, HBYTES, stream);
    k_edge_gather_norm<<<cd(BE * C_, 256), 256, 0, stream>>>(hX, src, dst, E_, m, dis, hY);
    k_gcn_combine<<<cd((long long)B_ * n * C_, 256), 256, 0, stream>>>(hX, hY, dis, bg, n);
  };

  gcn(hZ, k2, Wg3, bg3, srcB, dstB, mB);                              // -> hX
  pool(hX, k2, k3, p3, srcB, dstB, E_, mB, hZ, srcA, dstA, mA);       // edges B -> A
  gcn(hZ, k3, Wg4, bg4, srcA, dstA, mA);                              // -> hX
  pool(hX, k3, k4, p4, nullptr, nullptr, 0, nullptr, hZ, nullptr, nullptr, nullptr); // no remap

  k_head<<<B_, 128, 0, stream>>>(feats, action, W1, b1, W2, b2, W3, b3, out);
}
